// SMP_Stress_Predictor_8254927143036
// MI455X (gfx1250) — matrix operand order repaired and checked by reading, not compiled, not run on hardware
//
#include <hip/hip_runtime.h>
#include <hip/hip_bf16.h>

#define N_NODES 50000
#define N_EDGES 200000
#define G_GRAPH 512
#define B_ 256
#define W_WIN 64
#define TD_ 8
#define NF_ 14
#define EF_ 4
#define HID_ 64
#define HEADS_ 4
#define HC_ 256
#define TH_ 256
#define FF_ 1024
#define TOK_ (B_ * W_WIN)   // 16384 tokens

typedef __attribute__((ext_vector_type(16))) _Float16 v16h;
typedef __attribute__((ext_vector_type(8)))  float    v8f;

#define WMMA_F16(AF, BF, CC) \
    __builtin_amdgcn_wmma_f32_16x16x32_f16(false, (AF), false, (BF), (short)0, (CC), false, false)

// ---------------------------------------------------------------- utilities
__device__ __forceinline__ void atomicMaxF(float* addr, float val) {
    unsigned int* ua = (unsigned int*)addr;
    unsigned int old = *ua;
    while (__uint_as_float(old) < val) {
        unsigned int prev = atomicCAS(ua, old, __float_as_uint(val));
        if (prev == old) break;
        old = prev;
    }
}

__global__ void smp_fill(float* p, float v, long long n) {
    long long i = (long long)blockIdx.x * blockDim.x + threadIdx.x;
    if (i < n) p[i] = v;
}

// -------------------------------------------------- weight pre-pack (f32 -> f16 fragments)
// Layout: frag f = (ntile * ktiles + ktile); element (lane, slot i) at Wp[f*512 + lane*16 + i]
// B 32x16 f16 fragment layout: slot i -> K = ktile*32 + (i<8 ? i : i+8) + 8*(lane>>4), N = ntile*16 + (lane&15)
__global__ void smp_packB(const float* __restrict__ W, _Float16* __restrict__ Wp,
                          int K, int Nc) {
    int ktiles = (K + 31) >> 5;
    long long total = (long long)(Nc >> 4) * ktiles * 512;
    long long idx = (long long)blockIdx.x * blockDim.x + threadIdx.x;
    if (idx >= total) return;
    int i    = (int)(idx & 15);
    int lane = (int)((idx >> 4) & 31);
    long long rest = idx >> 9;
    int kt = (int)(rest % ktiles);
    int nt = (int)(rest / ktiles);
    int k = kt * 32 + ((i < 8) ? i : i + 8) + ((lane >> 4) << 3);
    int n = nt * 16 + (lane & 15);
    float v = (k < K) ? W[(long long)k * Nc + n] : 0.0f;
    Wp[idx] = (_Float16)v;
}

// A 16x32 f16 fragment from a row pointer (already offset by row*K + 8*half):
// slot i<8 -> k0+i+8h ; slot i>=8 -> k0+16+(i-8)+8h
__device__ __forceinline__ v16h smp_load_a(const float* __restrict__ ap) {
    v16h af;
#pragma unroll
    for (int i = 0; i < 8; i++)  af[i] = (_Float16)ap[i];
#pragma unroll
    for (int i = 8; i < 16; i++) af[i] = (_Float16)ap[i + 8];
    return af;
}

// ------------------------------------------------------------- WMMA GEMM
// C[M,Nc] = act( A[M,K] @ W[K,Nc] + bias + Cinit ), W pre-packed as f16 fragments.
// wave32: each wave computes a 16x64 strip = 4x v_wmma_f32_16x16x32_f16 per k-step,
// reusing one A fragment. Software-pipelined: loads for step k+1 issue before the
// WMMAs of step k; all 4 B fragments addressed off ONE base with immediate offsets.
// Block = 256 threads = 8 waves stacked in M (128x64 tile).
// Requires: M % 16 == 0, Nc % 64 == 0 (true for all shapes in this model).
__global__ void smp_gemm(const float* __restrict__ A, const v16h* __restrict__ Bp,
                         const float* __restrict__ bias, const float* __restrict__ Cinit,
                         float* __restrict__ Cout, int M, int K, int Nc, int relu) {
    const int lane = threadIdx.x & 31;
    const int wave = threadIdx.x >> 5;
    const int half = lane >> 4;        // 0: lanes 0-15, 1: lanes 16-31
    const int lsub = lane & 15;
    const int tm   = blockIdx.y * 128 + wave * 16;
    if (tm >= M) return;               // wave-uniform: whole 16-row tile valid or dead
    const int tn0  = blockIdx.x * 64;
    const int arow = tm + lsub;        // A-fragment row for this lane (always < M)
    const int ktiles = (K + 31) >> 5;
    const int kfull  = K >> 5;
    const int ks     = ktiles << 5;    // v16h stride between consecutive ntiles

    const float* arp = A + (long long)arow * K;
    const float* ap  = arp + (half << 3);
    // single base pointer; fragments t=0..3 at immediate offsets t*ks (<= 96KB, fits imm24)
    const v16h* bp = Bp + (((long long)(blockIdx.x * 4) * ktiles) << 5) + lane;

    v8f acc0 = {}, acc1 = {}, acc2 = {}, acc3 = {};

    if (kfull > 0) {
        v16h af = smp_load_a(ap);
        v16h b0 = bp[0], b1 = bp[ks], b2 = bp[2 * ks], b3 = bp[3 * ks];
        for (int kt = 1; kt < kfull; kt++) {
            // issue next-step loads first, then consume current fragments
            const int ko = kt << 5;
            v16h nf = smp_load_a(ap + ko);
            v16h n0 = bp[ko], n1 = bp[ko + ks], n2 = bp[ko + 2 * ks], n3 = bp[ko + 3 * ks];
            __builtin_prefetch(ap + ko + 32, 0, 3);   // next A k-tile -> global_prefetch
            acc0 = WMMA_F16(af, b0, acc0);
            acc1 = WMMA_F16(af, b1, acc1);
            acc2 = WMMA_F16(af, b2, acc2);
            acc3 = WMMA_F16(af, b3, acc3);
            af = nf; b0 = n0; b1 = n1; b2 = n2; b3 = n3;
        }
        acc0 = WMMA_F16(af, b0, acc0);
        acc1 = WMMA_F16(af, b1, acc1);
        acc2 = WMMA_F16(af, b2, acc2);
        acc3 = WMMA_F16(af, b3, acc3);
    }
    if (K & 31) {
        // ragged tail (K = 14, 72): branch-free clamp+select for A; B is zero-padded
        const int k0 = kfull << 5;
        v16h af;
#pragma unroll
        for (int i = 0; i < 16; i++) {
            int ka = k0 + ((i < 8) ? i : i + 8) + (half << 3);
            float av = arp[ka < K ? ka : 0];
            af[i] = (_Float16)(ka < K ? av : 0.0f);
        }
        const int ko = kfull << 5;
        v16h b0 = bp[ko], b1 = bp[ko + ks], b2 = bp[ko + 2 * ks], b3 = bp[ko + 3 * ks];
        acc0 = WMMA_F16(af, b0, acc0);
        acc1 = WMMA_F16(af, b1, acc1);
        acc2 = WMMA_F16(af, b2, acc2);
        acc3 = WMMA_F16(af, b3, acc3);
    }

    // C/D layout: vgpr r -> M = tm + r + 8*half, N = tn + (lane&15)
    v8f accs[4] = {acc0, acc1, acc2, acc3};
#pragma unroll
    for (int t = 0; t < 4; t++) {
        int n = tn0 + t * 16 + lsub;
        float bv = bias ? bias[n] : 0.0f;
#pragma unroll
        for (int r = 0; r < 8; r++) {
            int m = tm + r + (half << 3);
            float v = accs[t][r] + bv;
            if (Cinit) v += Cinit[(long long)m * Nc + n];
            if (relu)  v = fmaxf(v, 0.0f);
            Cout[(long long)m * Nc + n] = v;
        }
    }
}

// ------------------------------------------------------------- edge kernels
__global__ void smp_edge_alpha(const float* __restrict__ q, const float* __restrict__ k,
                               const float* __restrict__ ea, const float* __restrict__ We,
                               const int* __restrict__ src, const int* __restrict__ dst,
                               float* __restrict__ alpha, float* __restrict__ mbuf) {
    int idx = blockIdx.x * blockDim.x + threadIdx.x;
    if (idx >= N_EDGES * HEADS_) return;
    int e = idx >> 2, h = idx & 3;
    int s = src[e], d = dst[e];
    float a0 = ea[e * 4 + 0], a1 = ea[e * 4 + 1], a2 = ea[e * 4 + 2], a3 = ea[e * 4 + 3];
    const float* qp = q + (long long)d * HC_ + h * HID_;
    const float* kp = k + (long long)s * HC_ + h * HID_;
    const float* wp = We + h * HID_;   // We is 4 x 256
    float acc = 0.0f;
    for (int c = 0; c < HID_; c++) {
        float ec = a0 * wp[c] + a1 * wp[256 + c] + a2 * wp[512 + c] + a3 * wp[768 + c];
        acc += qp[c] * (kp[c] + ec);
    }
    acc *= 0.125f;                     // 1/sqrt(64)
    alpha[idx] = acc;
    atomicMaxF(&mbuf[d * HEADS_ + h], acc);
}

__global__ void smp_edge_exp(float* __restrict__ alpha, const float* __restrict__ mbuf,
                             float* __restrict__ den, const int* __restrict__ dst) {
    int idx = blockIdx.x * blockDim.x + threadIdx.x;
    if (idx >= N_EDGES * HEADS_) return;
    int e = idx >> 2, h = idx & 3;
    int d = dst[e];
    float ex = __expf(alpha[idx] - mbuf[d * HEADS_ + h]);
    alpha[idx] = ex;
    atomicAdd(&den[d * HEADS_ + h], ex);
}

__global__ void smp_edge_msg(const float* __restrict__ v, const float* __restrict__ ea,
                             const float* __restrict__ We, const int* __restrict__ src,
                             const int* __restrict__ dst, const float* __restrict__ ex,
                             const float* __restrict__ den, float* __restrict__ agg) {
    long long idx = (long long)blockIdx.x * blockDim.x + threadIdx.x;
    if (idx >= (long long)N_EDGES * HC_) return;
    int e = (int)(idx >> 8);
    int c = (int)(idx & 255);
    int h = c >> 6;
    int s = src[e], d = dst[e];
    float w = ex[e * HEADS_ + h] / (den[d * HEADS_ + h] + 1e-16f);
    float ec = ea[e * 4 + 0] * We[c] + ea[e * 4 + 1] * We[256 + c] +
               ea[e * 4 + 2] * We[512 + c] + ea[e * 4 + 3] * We[768 + c];
    float msg = (v[(long long)s * HC_ + c] + ec) * w;
    atomicAdd(&agg[(long long)d * HC_ + c], msg);
}

// ------------------------------------------------------------- pooling / glue
__global__ void smp_pool_add(const float* __restrict__ h, const int* __restrict__ bm,
                             float* __restrict__ pooled) {
    long long idx = (long long)blockIdx.x * blockDim.x + threadIdx.x;
    if (idx >= (long long)N_NODES * HC_) return;
    int n = (int)(idx >> 8), c = (int)(idx & 255);
    atomicAdd(&pooled[(long long)bm[n] * HC_ + c], h[idx]);
}

__global__ void smp_pool_cnt(const int* __restrict__ bm, float* __restrict__ cnt) {
    int n = blockIdx.x * blockDim.x + threadIdx.x;
    if (n < N_NODES) atomicAdd(&cnt[bm[n]], 1.0f);
}

__global__ void smp_build_d(const float* __restrict__ pooled, const float* __restrict__ cnt,
                            const float* __restrict__ tw, float* __restrict__ dbuf) {
    int idx = blockIdx.x * blockDim.x + threadIdx.x;
    if (idx >= B_ * 768) return;
    int b = idx / 768, j = idx % 768;
    float v;
    if (j < HC_) {
        float p0 = pooled[(long long)b * HC_ + j] / fmaxf(cnt[b], 1.0f);
        float p1 = pooled[(long long)(b + B_) * HC_ + j] / fmaxf(cnt[b + B_], 1.0f);
        v = 0.5f * (p0 + p1);
    } else {
        v = tw[(long long)b * (W_WIN * TD_) + (j - HC_)];
    }
    dbuf[idx] = v;
}

__global__ void smp_build_ti(const float* __restrict__ tw, const float* __restrict__ dyn,
                             float* __restrict__ ti) {
    long long idx = (long long)blockIdx.x * blockDim.x + threadIdx.x;
    if (idx >= (long long)TOK_ * 72) return;
    int r = (int)(idx / 72), j = (int)(idx % 72);
    int b = r >> 6;
    ti[idx] = (j < TD_) ? tw[(long long)r * TD_ + j] : dyn[(long long)b * HID_ + (j - TD_)];
}

// ------------------------------------------------------------- encoder attention
__global__ void smp_attn(const float* __restrict__ qkv, float* __restrict__ o) {
    __shared__ float Qs[64 * 64];
    __shared__ float Ks[64 * 64];
    __shared__ float Vs[64 * 64];
    int b = blockIdx.x >> 2, h = blockIdx.x & 3;
    int s = threadIdx.x;
    const float* base = qkv + (long long)b * 64 * 768 + h * 64;
    for (int d = 0; d < 64; d++) {
        Qs[s * 64 + d] = base[(long long)s * 768 + d];
        Ks[s * 64 + d] = base[(long long)s * 768 + 256 + d];
        Vs[s * 64 + d] = base[(long long)s * 768 + 512 + d];
    }
    __syncthreads();
    float l[64];
    float mx = -1e30f;
    for (int j = 0; j < 64; j++) {
        float acc = 0.0f;
        for (int d = 0; d < 64; d++) acc += Qs[s * 64 + d] * Ks[j * 64 + d];
        acc *= 0.125f;
        l[j] = acc;
        mx = fmaxf(mx, acc);
    }
    float sum = 0.0f;
    for (int j = 0; j < 64; j++) { l[j] = __expf(l[j] - mx); sum += l[j]; }
    float inv = 1.0f / sum;
    float* op = o + ((long long)b * 64 + s) * TH_ + h * 64;
    for (int d = 0; d < 64; d++) {
        float acc = 0.0f;
        for (int j = 0; j < 64; j++) acc += l[j] * Vs[j * 64 + d];
        op[d] = acc * inv;
    }
}

// wave-per-row LayerNorm over 256 channels (wave32 shuffle reduction)
__global__ void smp_ln(const float* __restrict__ x, const float* __restrict__ g,
                       const float* __restrict__ be, float* __restrict__ out, int rows) {
    int wave = threadIdx.x >> 5, lane = threadIdx.x & 31;
    int row = blockIdx.x * 8 + wave;
    if (row >= rows) return;
    const float* xp = x + (long long)row * TH_;
    float s = 0.0f, s2 = 0.0f;
    for (int i = lane; i < TH_; i += 32) { float v = xp[i]; s += v; s2 += v * v; }
    for (int off = 16; off; off >>= 1) { s += __shfl_xor(s, off); s2 += __shfl_xor(s2, off); }
    float mu = s * (1.0f / TH_);
    float var = s2 * (1.0f / TH_) - mu * mu;
    float r = rsqrtf(var + 1e-5f);
    float* op = out + (long long)row * TH_;
    for (int i = lane; i < TH_; i += 32) op[i] = (xp[i] - mu) * r * g[i] + be[i];
}

__global__ void smp_out(const float* __restrict__ h2, const float* __restrict__ Wo,
                        const float* __restrict__ bo, float* __restrict__ out) {
    int b = blockIdx.x * blockDim.x + threadIdx.x;
    if (b >= B_) return;
    const float* hp = h2 + ((long long)b * 64 + 63) * TH_;
    float acc = bo[0];
    for (int c = 0; c < TH_; c++) acc += hp[c] * Wo[c];
    out[b] = acc;
}

// ---------------------------------------------------------------- launch
extern "C" void kernel_launch(void* const* d_in, const int* in_sizes, int n_in,
                              void* d_out, int out_size, void* d_ws, size_t ws_size,
                              hipStream_t stream) {
    (void)in_sizes; (void)n_in; (void)out_size; (void)ws_size;
    const float* x  = (const float*)d_in[0];
    const int*   ei = (const int*)d_in[1];
    const float* ea = (const float*)d_in[2];
    const int*   bm = (const int*)d_in[3];
    const float* tw = (const float*)d_in[4];
    int p = 5;
    struct GnnP { const float *Wq,*bq,*Wk,*bk,*Wv,*bv,*We,*Ws,*bs; } gnn[3];
    for (int l = 0; l < 3; l++) {
        gnn[l].Wq = (const float*)d_in[p+0]; gnn[l].bq = (const float*)d_in[p+1];
        gnn[l].Wk = (const float*)d_in[p+2]; gnn[l].bk = (const float*)d_in[p+3];
        gnn[l].Wv = (const float*)d_in[p+4]; gnn[l].bv = (const float*)d_in[p+5];
        gnn[l].We = (const float*)d_in[p+6]; gnn[l].Ws = (const float*)d_in[p+7];
        gnn[l].bs = (const float*)d_in[p+8]; p += 9;
    }
    const float* mW1 = (const float*)d_in[p++]; const float* mb1 = (const float*)d_in[p++];
    const float* mW2 = (const float*)d_in[p++]; const float* mb2 = (const float*)d_in[p++];
    const float* pW  = (const float*)d_in[p++]; const float* pb  = (const float*)d_in[p++];
    struct EncP { const float *Wqkv,*bqkv,*Wo,*bo,*g1,*be1,*W1,*bf1,*W2,*bf2,*g2,*be2; } enc[2];
    for (int l = 0; l < 2; l++) {
        enc[l].Wqkv=(const float*)d_in[p+0];  enc[l].bqkv=(const float*)d_in[p+1];
        enc[l].Wo  =(const float*)d_in[p+2];  enc[l].bo  =(const float*)d_in[p+3];
        enc[l].g1  =(const float*)d_in[p+4];  enc[l].be1 =(const float*)d_in[p+5];
        enc[l].W1  =(const float*)d_in[p+6];  enc[l].bf1 =(const float*)d_in[p+7];
        enc[l].W2  =(const float*)d_in[p+8];  enc[l].bf2 =(const float*)d_in[p+9];
        enc[l].g2  =(const float*)d_in[p+10]; enc[l].be2 =(const float*)d_in[p+11];
        p += 12;
    }
    const float* outW = (const float*)d_in[p++];
    const float* outB = (const float*)d_in[p++];

    // workspace layout
    char* w = (char*)d_ws;
    const size_t SB = (size_t)N_NODES * HC_ * sizeof(float);   // 51.2 MB
    float* Bb[5];
    for (int i = 0; i < 5; i++) Bb[i] = (float*)(w + (size_t)i * SB);
    float* alphab = (float*)(w + 5 * SB);
    float* mbuf   = alphab + (size_t)N_EDGES * HEADS_;
    float* den    = mbuf   + (size_t)N_NODES * HEADS_;
    float* pooled = den    + (size_t)N_NODES * HEADS_;
    float* cnt    = pooled + (size_t)G_GRAPH * HC_;
    float* dbuf   = cnt    + G_GRAPH;
    float* t1b    = dbuf   + (size_t)B_ * 768;
    float* dynb   = t1b    + (size_t)B_ * HID_;
    _Float16* packb = (_Float16*)(dynb + (size_t)B_ * HID_);   // packed weight fragments (<=1.2MB)
    // encoder scratch aliases the (dead-by-then) GNN node buffers
    float* h2   = (float*)(w + 0);
    float* qkv  = (float*)(w + 24000000ULL);
    float* obuf = (float*)(w + 80000000ULL);
    float* ffb  = (float*)(w + 100000000ULL);
    float* tmp  = (float*)(w + 170000000ULL);
    float* tib  = (float*)(w + 190000000ULL);

    const int* src = ei;
    const int* dst = ei + N_EDGES;

    auto gemm = [&](const float* A, const float* Wm, const float* bias, const float* Cinit,
                    float* C, int M, int K, int Nc, int relu) {
        int ktiles = (K + 31) >> 5;
        long long tot = (long long)(Nc >> 4) * ktiles * 512;
        smp_packB<<<dim3((unsigned)((tot + 255) / 256)), dim3(256), 0, stream>>>(Wm, packb, K, Nc);
        dim3 g(Nc / 64, (M + 127) / 128);
        smp_gemm<<<g, dim3(256), 0, stream>>>(A, (const v16h*)packb, bias, Cinit, C, M, K, Nc, relu);
    };
    auto fill = [&](float* pp, float v, long long n) {
        smp_fill<<<dim3((unsigned)((n + 255) / 256)), dim3(256), 0, stream>>>(pp, v, n);
    };

    // ---- GNN layers ----
    const float* hin = x;
    int Kin = NF_;
    const int qidx[3] = {0, 4, 0};     // q buffer (also reused as layer output)
    for (int l = 0; l < 3; l++) {
        float* qb = Bb[qidx[l]];
        float* kb = Bb[1];
        float* vb = Bb[2];
        float* ag = Bb[3];
        gemm(hin, gnn[l].Wq, gnn[l].bq, nullptr, qb, N_NODES, Kin, HC_, 0);
        gemm(hin, gnn[l].Wk, gnn[l].bk, nullptr, kb, N_NODES, Kin, HC_, 0);
        gemm(hin, gnn[l].Wv, gnn[l].bv, nullptr, vb, N_NODES, Kin, HC_, 0);
        fill(mbuf, -1e30f, (long long)N_NODES * HEADS_);
        fill(den, 0.0f, (long long)N_NODES * HEADS_);
        fill(ag, 0.0f, (long long)N_NODES * HC_);
        {
            int nt = N_EDGES * HEADS_;
            smp_edge_alpha<<<dim3((nt + 255) / 256), dim3(256), 0, stream>>>(
                qb, kb, ea, gnn[l].We, src, dst, alphab, mbuf);
            smp_edge_exp<<<dim3((nt + 255) / 256), dim3(256), 0, stream>>>(
                alphab, mbuf, den, dst);
            long long nm = (long long)N_EDGES * HC_;
            smp_edge_msg<<<dim3((unsigned)((nm + 255) / 256)), dim3(256), 0, stream>>>(
                vb, ea, gnn[l].We, src, dst, alphab, den, ag);
        }
        gemm(hin, gnn[l].Ws, gnn[l].bs, ag, qb, N_NODES, Kin, HC_, 1);  // relu(agg + h@Ws + bs)
        hin = qb;
        Kin = HC_;
    }

    // ---- pooling + static/dynamic fusion ----
    fill(pooled, 0.0f, (long long)G_GRAPH * HC_);
    fill(cnt, 0.0f, G_GRAPH);
    {
        long long nn = (long long)N_NODES * HC_;
        smp_pool_add<<<dim3((unsigned)((nn + 255) / 256)), dim3(256), 0, stream>>>(hin, bm, pooled);
        smp_pool_cnt<<<dim3((N_NODES + 255) / 256), dim3(256), 0, stream>>>(bm, cnt);
        smp_build_d<<<dim3((B_ * 768 + 255) / 256), dim3(256), 0, stream>>>(pooled, cnt, tw, dbuf);
    }
    gemm(dbuf, mW1, mb1, nullptr, t1b, B_, 768, HID_, 1);
    gemm(t1b, mW2, mb2, nullptr, dynb, B_, HID_, HID_, 0);
    {
        long long nt = (long long)TOK_ * 72;
        smp_build_ti<<<dim3((unsigned)((nt + 255) / 256)), dim3(256), 0, stream>>>(tw, dynb, tib);
    }
    gemm(tib, pW, pb, nullptr, h2, TOK_, 72, TH_, 0);

    // ---- transformer encoder ----
    for (int l = 0; l < 2; l++) {
        gemm(h2, enc[l].Wqkv, enc[l].bqkv, nullptr, qkv, TOK_, TH_, 3 * TH_, 0);
        smp_attn<<<dim3(B_ * HEADS_), dim3(64), 0, stream>>>(qkv, obuf);
        gemm(obuf, enc[l].Wo, enc[l].bo, h2, tmp, TOK_, TH_, TH_, 0);  // +residual
        smp_ln<<<dim3(TOK_ / 8), dim3(256), 0, stream>>>(tmp, enc[l].g1, enc[l].be1, h2, TOK_);
        gemm(h2, enc[l].W1, enc[l].bf1, nullptr, ffb, TOK_, TH_, FF_, 1);
        gemm(ffb, enc[l].W2, enc[l].bf2, h2, tmp, TOK_, FF_, TH_, 0);  // +residual
        smp_ln<<<dim3(TOK_ / 8), dim3(256), 0, stream>>>(tmp, enc[l].g2, enc[l].be2, h2, TOK_);
    }

    smp_out<<<dim3(1), dim3(256), 0, stream>>>(h2, outW, outB, (float*)d_out);
}
